// TokenEvictionLayer_15805479649733
// MI455X (gfx1250) — compile-verified, matrix operand order checked by reading, not run on hardware
//
#include <hip/hip_runtime.h>
#include <hip/hip_fp16.h>
#include <math.h>

typedef _Float16 v16h __attribute__((ext_vector_type(16)));
typedef float    v8f  __attribute__((ext_vector_type(8)));

#define DIM   1024
#define SEQ   8192
#define BATCH 4
#define HID   128
#define NKEEP 1024
#define TOK_PER_BLOCK 128   // 8 waves * 16 tokens
#define KCHUNK 64           // two 32-wide WMMA K steps per chunk

// Workspace layout (bytes):
//   [0, 128K)        : scores, float [BATCH*SEQ]
//   [128K, 144K)     : kept indices, int [BATCH*NKEEP]
//   [144K, 400K)     : W1 as f16 in WMMA B-fragment order [DIM*HID]
#define WS_SCORES_OFF 0
#define WS_IDX_OFF    ((size_t)BATCH * SEQ * sizeof(float))
#define WS_W1_OFF     (WS_IDX_OFF + (size_t)BATCH * NKEEP * sizeof(int))

// ---------------------------------------------------------------------------
// Prep kernel: swizzle W1 (f32, [k][col]) into f16 WMMA B-fragment layout:
//   dst halfword index = ((kstep*8 + ntile)*32 + lane)*16 + h
// where kstep=k/32, K=k%32, ntile=col/16, n=col%16,
//   hi = (K>>3)&1, lane = n + 16*hi, h = (K&7) + (K>=16 ? 8 : 0)
// This makes every B fragment a single contiguous 32-byte per-lane read.
// ---------------------------------------------------------------------------
__global__ __launch_bounds__(256)
void te_prep_w1_kernel(const float* __restrict__ W1, _Float16* __restrict__ w1f) {
  const int idx = blockIdx.x * 256 + threadIdx.x;   // 0 .. DIM*HID-1
  const int k   = idx >> 7;                         // row (K dim)
  const int col = idx & 127;                        // hidden column
  const int ks  = k >> 5;                           // 32-wide K step
  const int K   = k & 31;
  const int t   = col >> 4;
  const int n   = col & 15;
  const int hi  = (K >> 3) & 1;
  const int h   = (K & 7) + ((K >= 16) ? 8 : 0);
  const int lane = n + (hi << 4);
  w1f[(((size_t)ks * 8 + t) * 32 + lane) * 16 + h] = (_Float16)W1[idx];
}

// ---------------------------------------------------------------------------
// Kernel 1: fused importance scorer.
// h = GELU(x @ W1 + b1); scores = h @ W2 + b2
// Each wave32 computes a 16-token x 128-hidden tile with
// v_wmma_f32_16x16x32_f16 (8 N-tiles, K swept in 32-wide steps).
// B fragments staged in LDS pre-swizzled -> ds_load_b128 reads only.
// ---------------------------------------------------------------------------
__global__ __launch_bounds__(256)
void te_score_kernel(const float* __restrict__ x,
                     const _Float16* __restrict__ w1f,   // swizzled f16 W1
                     const float* __restrict__ b1, const float* __restrict__ W2,
                     const float* __restrict__ b2, float* __restrict__ scores) {
  __shared__ _Float16 lwB[2 * 8 * 32 * 16];       // 16 KB: one 64-K chunk

  const int tid    = threadIdx.x;
  const int lane   = tid & 31;
  const int wave   = tid >> 5;
  const int blocksPerBatch = SEQ / TOK_PER_BLOCK; // 64
  const int b      = blockIdx.x / blocksPerBatch;
  const int blk    = blockIdx.x % blocksPerBatch;
  const int token0 = blk * TOK_PER_BLOCK + wave * 16;

  const int m      = lane & 15;        // M row (A) / N col (B,C) within tile
  const int hiHalf = lane >> 4;
  const int koff   = hiHalf * 8;       // ISA 16-bit A layout K offset

  const float* xrow = x + ((size_t)b * SEQ + token0 + m) * (size_t)DIM;
  const v16h*  lwv  = (const v16h*)lwB;

  v8f acc[8] = {};                     // 16x128 f32 accumulators per wave

  for (int kc = 0; kc < DIM; kc += KCHUNK) {
    // contiguous 16 KB copy: swizzled W1 chunk -> LDS (b128 traffic only)
    {
      const uint4* src = (const uint4*)(w1f + (size_t)kc * HID);
      uint4* dst = (uint4*)lwB;
      #pragma unroll
      for (int i = 0; i < 4; ++i)
        dst[tid + i * 256] = src[tid + i * 256];
    }
    __syncthreads();

    // prefetch next chunk of this token's row (gfx1250 global_prefetch_b8)
    if (kc + KCHUNK < DIM)
      __builtin_prefetch(xrow + kc + KCHUNK + koff, 0, 1);

    #pragma unroll
    for (int s = 0; s < 2; ++s) {                 // two 32-wide K steps
      const int kb = kc + s * 32;
      // A fragment: lane's halves live at K = kb+koff+[0,8) and +16+[0,8)
      const float4 p0 = *(const float4*)(xrow + kb + koff);
      const float4 p1 = *(const float4*)(xrow + kb + koff + 4);
      const float4 p2 = *(const float4*)(xrow + kb + koff + 16);
      const float4 p3 = *(const float4*)(xrow + kb + koff + 20);
      v16h a;
      a[0]  = (_Float16)p0.x; a[1]  = (_Float16)p0.y;
      a[2]  = (_Float16)p0.z; a[3]  = (_Float16)p0.w;
      a[4]  = (_Float16)p1.x; a[5]  = (_Float16)p1.y;
      a[6]  = (_Float16)p1.z; a[7]  = (_Float16)p1.w;
      a[8]  = (_Float16)p2.x; a[9]  = (_Float16)p2.y;
      a[10] = (_Float16)p2.z; a[11] = (_Float16)p2.w;
      a[12] = (_Float16)p3.x; a[13] = (_Float16)p3.y;
      a[14] = (_Float16)p3.z; a[15] = (_Float16)p3.w;

      #pragma unroll
      for (int t = 0; t < 8; ++t) {               // 8 N-tiles -> 128 hidden
        const v16h bf = lwv[(s * 8 + t) * 32 + lane];   // one 32B LDS read
        acc[t] = __builtin_amdgcn_wmma_f32_16x16x32_f16(
            false, a, false, bf, (short)0, acc[t], false, false);
      }
    }
    __syncthreads();
  }

  // Epilogue: bias + exact GELU + dot with W2, reduce across the N dimension.
  // C/D layout: element r of acc[t] is h[M = r + 8*hiHalf][N = t*16 + m].
  float part[8];
  #pragma unroll
  for (int r = 0; r < 8; ++r) part[r] = 0.0f;

  #pragma unroll
  for (int t = 0; t < 8; ++t) {
    const int n     = t * 16 + m;
    const float w   = W2[n];
    const float bb1 = b1[n];
    #pragma unroll
    for (int r = 0; r < 8; ++r) {
      float hv = acc[t][r] + bb1;
      hv = 0.5f * hv * (1.0f + erff(hv * 0.70710678118654752f));  // exact GELU
      part[r] += hv * w;
    }
  }
  #pragma unroll
  for (int r = 0; r < 8; ++r)
    for (int off = 1; off < 16; off <<= 1)
      part[r] += __shfl_xor(part[r], off, 16);

  if (m == 0) {                                   // lanes 0 and 16 write
    const float bb2 = b2[0];
    const int mbase = hiHalf * 8;
    #pragma unroll
    for (int r = 0; r < 8; ++r)
      scores[(size_t)b * SEQ + token0 + mbase + r] = part[r] + bb2;
  }
}

// ---------------------------------------------------------------------------
// Kernel 2: per-batch top-1024 via full bitonic sort of packed (score, idx)
// keys in LDS, then ascending bitonic sort of the kept indices.
// Key = monotone(score) << 13 | (8191 - idx)  => descending sort keeps the
// largest scores, lowest index first on ties (matches jax.lax.top_k).
// ---------------------------------------------------------------------------
__global__ __launch_bounds__(1024)
void te_topk_kernel(const float* __restrict__ scores, int* __restrict__ idxOut,
                    float* __restrict__ outF) {
  __shared__ unsigned long long keys[SEQ];        // 64 KB
  const int tid = threadIdx.x;
  const int b   = blockIdx.x;

  for (int i = tid; i < SEQ; i += 1024) {
    const float s = scores[(size_t)b * SEQ + i];
    unsigned u = __float_as_uint(s);
    u = (u & 0x80000000u) ? ~u : (u | 0x80000000u);       // order-preserving
    keys[i] = ((unsigned long long)u << 13) |
              (unsigned long long)(SEQ - 1 - i);
  }
  __syncthreads();

  // descending bitonic sort, 8192 elements, 8 slots per thread
  for (int k = 2; k <= SEQ; k <<= 1) {
    for (int j = k >> 1; j > 0; j >>= 1) {
      #pragma unroll
      for (int e = 0; e < SEQ / 1024; ++e) {
        const int i   = e * 1024 + tid;
        const int ixj = i ^ j;
        if (ixj > i) {
          const unsigned long long va = keys[i];
          const unsigned long long vb = keys[ixj];
          const bool desc = ((i & k) == 0);
          if (desc ? (va < vb) : (va > vb)) { keys[i] = vb; keys[ixj] = va; }
        }
      }
      __syncthreads();
    }
  }

  // extract surviving indices, then sort them ascending (restore time order)
  const int myidx = (SEQ - 1) - (int)(keys[tid] & (unsigned long long)(SEQ - 1));
  __syncthreads();
  int* sidx = (int*)keys;                         // reuse LDS
  sidx[tid] = myidx;
  __syncthreads();

  for (int k = 2; k <= NKEEP; k <<= 1) {
    for (int j = k >> 1; j > 0; j >>= 1) {
      const int i = tid, ixj = i ^ j;
      if (ixj > i) {
        const int va = sidx[i], vb = sidx[ixj];
        const bool asc = ((i & k) == 0);
        if (asc ? (va > vb) : (va < vb)) { sidx[i] = vb; sidx[ixj] = va; }
      }
      __syncthreads();
    }
  }

  const int v = sidx[tid];
  idxOut[b * NKEEP + tid] = v;                    // for the gather kernel
  // second tuple output: indices, flattened after kept_tokens
  outF[(size_t)BATCH * NKEEP * DIM + (size_t)b * NKEEP + tid] = (float)v;
}

// ---------------------------------------------------------------------------
// Kernel 3: gather kept rows. One block per kept token, float4 copies.
// ---------------------------------------------------------------------------
__global__ __launch_bounds__(256)
void te_gather_kernel(const float* __restrict__ x, const int* __restrict__ idx,
                      float* __restrict__ out) {
  const int b = blockIdx.x / NKEEP;
  const int r = blockIdx.x % NKEEP;
  const int srow = idx[b * NKEEP + r];
  const float4* __restrict__ src =
      (const float4*)(x + ((size_t)b * SEQ + srow) * (size_t)DIM);
  float4* __restrict__ dst =
      (float4*)(out + ((size_t)b * NKEEP + r) * (size_t)DIM);
  dst[threadIdx.x] = src[threadIdx.x];            // 256 * 16B = 4 KB row
}

// ---------------------------------------------------------------------------
extern "C" void kernel_launch(void* const* d_in, const int* in_sizes, int n_in,
                              void* d_out, int out_size, void* d_ws, size_t ws_size,
                              hipStream_t stream) {
  const float* x  = (const float*)d_in[0];
  const float* W1 = (const float*)d_in[1];
  const float* b1 = (const float*)d_in[2];
  const float* W2 = (const float*)d_in[3];
  const float* b2 = (const float*)d_in[4];
  float* out = (float*)d_out;

  float*    scores = (float*)((char*)d_ws + WS_SCORES_OFF);
  int*      idxWs  = (int*)((char*)d_ws + WS_IDX_OFF);
  _Float16* w1f    = (_Float16*)((char*)d_ws + WS_W1_OFF);

  te_prep_w1_kernel<<<(DIM * HID) / 256, 256, 0, stream>>>(W1, w1f);
  te_score_kernel<<<BATCH * (SEQ / TOK_PER_BLOCK), 256, 0, stream>>>(
      x, w1f, b1, W2, b2, scores);
  te_topk_kernel<<<BATCH, 1024, 0, stream>>>(scores, idxWs, out);
  te_gather_kernel<<<BATCH * NKEEP, 256, 0, stream>>>(x, idxWs, out);
}
